// PolySelfAttention_12068858102233
// MI455X (gfx1250) — compile-verified, matrix operand order checked by reading, not run on hardware
//
#include <hip/hip_runtime.h>
#include <hip/hip_bf16.h>

// ---------------------------------------------------------------------------
// PolySelfAttention on MI455X (gfx1250), bf16 WMMA pipeline.
//   B=8, N=4096, C=768, H=12, D=64, M=B*N=32768, 3C=2304
// ---------------------------------------------------------------------------

typedef __attribute__((ext_vector_type(16))) __bf16 v16bf;
typedef __attribute__((ext_vector_type(8)))  __bf16 v8bf;
typedef __attribute__((ext_vector_type(8)))  float  v8f;

static constexpr int kB = 8;
static constexpr int kN = 4096;
static constexpr int kC = 768;
static constexpr int kH = 12;
static constexpr int kD = 64;
static constexpr int kM = kB * kN;      // 32768 rows
static constexpr int kC3 = 3 * kC;      // 2304
static constexpr float kLnEps = 1e-5f;

// ---------------------------------------------------------------------------
// f32 -> bf16 conversion (x, w_qkv, w_proj)
// ---------------------------------------------------------------------------
__global__ __launch_bounds__(256) void cvt_f32_bf16(const float* __restrict__ in,
                                                    __bf16* __restrict__ out,
                                                    int n) {
  int i = blockIdx.x * 256 + threadIdx.x;
  if (i < n) out[i] = (__bf16)in[i];
}

// ---------------------------------------------------------------------------
// WMMA fragment load for 16x32 bf16 A/B tiles (row-major, ld = K).
// ISA layout (16-bit A 16x32): lanes 0-15 hold row M=lane, K={0..7,16..23};
// lanes 16-31 hold row M=lane-16, K={8..15,24..31}.  Two aligned b128 loads.
// ---------------------------------------------------------------------------
__device__ __forceinline__ v16bf ld_frag16x32(const __bf16* __restrict__ p) {
  v8bf lo = *(const v8bf*)(p);
  v8bf hi = *(const v8bf*)(p + 16);
  v16bf f;
#pragma unroll
  for (int i = 0; i < 8; ++i) { f[i] = lo[i]; f[i + 8] = hi[i]; }
  return f;
}

template <typename T> __device__ __forceinline__ T cvt_out(float v);
template <> __device__ __forceinline__ float  cvt_out<float >(float v) { return v; }
template <> __device__ __forceinline__ __bf16 cvt_out<__bf16>(float v) { return (__bf16)v; }

// ---------------------------------------------------------------------------
// C = A[M,K] * B[Nn,K]^T (+ bias), A/B bf16 row-major, f32 accumulate.
// Block = 256 threads = 8 waves arranged 4(M) x 2(N); each wave computes a
// 32x64 strip with 2 A-frags x 4 B-frags -> 8 wmma accumulators per K-step.
// Block tile = 128 x 128.  Requires M%128==0, Nn%128==0, K%32==0.
// ---------------------------------------------------------------------------
template <typename OutT, bool HAS_BIAS>
__global__ __launch_bounds__(256) void wmma_gemm_bf16(
    const __bf16* __restrict__ A, const __bf16* __restrict__ Bm,
    const float* __restrict__ bias, OutT* __restrict__ Cc,
    int M, int Nn, int K) {
  const int lane = threadIdx.x & 31;
  const int wave = threadIdx.x >> 5;
  const int m0 = blockIdx.y * 128 + (wave >> 1) * 32;  // 32 rows per wave
  const int n0 = blockIdx.x * 128 + (wave & 1) * 64;   // 64 cols per wave

  const int roff = lane & 15;              // row within 16-row fragment
  const int koff = (lane & 16) ? 8 : 0;    // K sub-block per half-wave

  const __bf16* arow0 = A + (size_t)(m0 + 0 * 16 + roff) * K + koff;
  const __bf16* arow1 = A + (size_t)(m0 + 1 * 16 + roff) * K + koff;
  const __bf16* brow0 = Bm + (size_t)(n0 + 0 * 16 + roff) * K + koff;
  const __bf16* brow1 = Bm + (size_t)(n0 + 1 * 16 + roff) * K + koff;
  const __bf16* brow2 = Bm + (size_t)(n0 + 2 * 16 + roff) * K + koff;
  const __bf16* brow3 = Bm + (size_t)(n0 + 3 * 16 + roff) * K + koff;

  v8f acc[2][4];
#pragma unroll
  for (int i = 0; i < 2; ++i)
#pragma unroll
    for (int j = 0; j < 4; ++j) acc[i][j] = (v8f){};

  for (int kb = 0; kb < K; kb += 32) {
    // prefetch next A chunks into near caches (global_prefetch_b8, WGP scope)
    __builtin_prefetch(arow0 + kb + 256, 0, 3);
    __builtin_prefetch(arow1 + kb + 256, 0, 3);
    v16bf a0 = ld_frag16x32(arow0 + kb);
    v16bf a1 = ld_frag16x32(arow1 + kb);
    v16bf b0 = ld_frag16x32(brow0 + kb);
    v16bf b1 = ld_frag16x32(brow1 + kb);
    v16bf b2 = ld_frag16x32(brow2 + kb);
    v16bf b3 = ld_frag16x32(brow3 + kb);
    acc[0][0] = __builtin_amdgcn_wmma_f32_16x16x32_bf16(false, a0, false, b0, (short)0, acc[0][0], false, false);
    acc[0][1] = __builtin_amdgcn_wmma_f32_16x16x32_bf16(false, a0, false, b1, (short)0, acc[0][1], false, false);
    acc[0][2] = __builtin_amdgcn_wmma_f32_16x16x32_bf16(false, a0, false, b2, (short)0, acc[0][2], false, false);
    acc[0][3] = __builtin_amdgcn_wmma_f32_16x16x32_bf16(false, a0, false, b3, (short)0, acc[0][3], false, false);
    acc[1][0] = __builtin_amdgcn_wmma_f32_16x16x32_bf16(false, a1, false, b0, (short)0, acc[1][0], false, false);
    acc[1][1] = __builtin_amdgcn_wmma_f32_16x16x32_bf16(false, a1, false, b1, (short)0, acc[1][1], false, false);
    acc[1][2] = __builtin_amdgcn_wmma_f32_16x16x32_bf16(false, a1, false, b2, (short)0, acc[1][2], false, false);
    acc[1][3] = __builtin_amdgcn_wmma_f32_16x16x32_bf16(false, a1, false, b3, (short)0, acc[1][3], false, false);
  }

  // D layout: lanes 0-15 => N=lane, rows r; lanes 16-31 => rows r+8.
  const int cn = lane & 15;
  const int mro = (lane & 16) ? 8 : 0;
#pragma unroll
  for (int j = 0; j < 4; ++j) {
    const int n = n0 + j * 16 + cn;
    const float bv = HAS_BIAS ? bias[n] : 0.0f;
#pragma unroll
    for (int i = 0; i < 2; ++i) {
#pragma unroll
      for (int r = 0; r < 8; ++r) {
        const int m = m0 + i * 16 + r + mro;
        Cc[(size_t)m * Nn + n] = cvt_out<OutT>(acc[i][j][r] + bv);
      }
    }
  }
}

// ---------------------------------------------------------------------------
// modulator[b,h,d] = LN_d( sum_n p2[h,n] * k[b,h,n,d] * v[b,h,n,d] )
// qkv is [M, 2304] bf16; k col = C + h*64+d, v col = 2C + h*64+d.
// One block per (b,h): 256 threads = 4 n-groups x 64 d.
// ---------------------------------------------------------------------------
__global__ __launch_bounds__(256) void modulator_ln_kernel(
    const __bf16* __restrict__ qkv, const float* __restrict__ p2,
    const float* __restrict__ gamma, const float* __restrict__ beta,
    float* __restrict__ modln) {
  const int bh = blockIdx.x;           // [0, B*H)
  const int b = bh / kH;
  const int h = bh % kH;
  const int d = threadIdx.x & 63;
  const int g = threadIdx.x >> 6;      // 0..3

  const __bf16* kbase = qkv + (size_t)b * kN * kC3 + kC + h * kD + d;
  const __bf16* vbase = kbase + kC;
  const float* p2h = p2 + (size_t)h * kN;

  float sum = 0.0f;
  for (int n = g; n < kN; n += 4) {
    const size_t o = (size_t)n * kC3;
    sum += p2h[n] * ((float)kbase[o] * (float)vbase[o]);
  }

  __shared__ float red[256];
  __shared__ float mv[2];
  red[threadIdx.x] = sum;
  __syncthreads();
  if (threadIdx.x < 64) {
    float tot = red[threadIdx.x] + red[threadIdx.x + 64] +
                red[threadIdx.x + 128] + red[threadIdx.x + 192];
    red[threadIdx.x] = tot;
  }
  __syncthreads();
  if (threadIdx.x == 0) {
    float mu = 0.0f;
    for (int i = 0; i < 64; ++i) mu += red[i];
    mu *= (1.0f / 64.0f);
    float var = 0.0f;
    for (int i = 0; i < 64; ++i) { float t = red[i] - mu; var += t * t; }
    var *= (1.0f / 64.0f);
    mv[0] = mu;
    mv[1] = rsqrtf(var + kLnEps);
  }
  __syncthreads();
  if (threadIdx.x < 64) {
    float v = (red[threadIdx.x] - mv[0]) * mv[1] * gamma[d] + beta[d];
    modln[(size_t)bh * kD + d] = v;
  }
}

// ---------------------------------------------------------------------------
// y[m, c] = q[m, c] * p1[h, n] * modln[b, h, d]   (c = h*64+d, m = b*N+n)
// written bf16 row-major [M, C] for the projection GEMM.
// ---------------------------------------------------------------------------
__global__ __launch_bounds__(256) void modulate_kernel(
    const __bf16* __restrict__ qkv, const float* __restrict__ p1,
    const float* __restrict__ modln, __bf16* __restrict__ y) {
  const size_t total = (size_t)kM * kC;
  size_t i = (size_t)blockIdx.x * 256 + threadIdx.x;
  if (i >= total) return;
  const int m = (int)(i / kC);
  const int c = (int)(i % kC);
  const int h = c >> 6;
  const int d = c & 63;
  const int b = m >> 12;       // /4096
  const int n = m & 4095;
  const float q = (float)qkv[(size_t)m * kC3 + c];
  const float val = q * p1[(size_t)h * kN + n] * modln[((size_t)(b * kH + h)) * kD + d];
  y[i] = (__bf16)val;
}

// ---------------------------------------------------------------------------
// launch
// ---------------------------------------------------------------------------
extern "C" void kernel_launch(void* const* d_in, const int* in_sizes, int n_in,
                              void* d_out, int out_size, void* d_ws, size_t ws_size,
                              hipStream_t stream) {
  const float* x      = (const float*)d_in[0];   // [B,N,C]
  const float* w_qkv  = (const float*)d_in[1];   // [3C,C]
  const float* w_proj = (const float*)d_in[2];   // [C,C]
  const float* b_proj = (const float*)d_in[3];   // [C]
  const float* p1     = (const float*)d_in[4];   // [H,N]
  const float* p2     = (const float*)d_in[5];   // [H,N]
  const float* gamma  = (const float*)d_in[6];   // [D]
  const float* beta   = (const float*)d_in[7];   // [D]
  float* out = (float*)d_out;                    // [B,N,C]

  (void)in_sizes; (void)n_in; (void)out_size; (void)ws_size;

  // workspace carve-up (all sizes are 256B multiples)
  char* ws = (char*)d_ws;
  size_t off = 0;
  auto carve = [&](size_t bytes) {
    void* p = ws + off;
    off += (bytes + 255) & ~(size_t)255;
    return p;
  };
  __bf16* xbf    = (__bf16*)carve((size_t)kM * kC * 2);     // 50.3 MB
  __bf16* wqkvbf = (__bf16*)carve((size_t)kC3 * kC * 2);    //  3.5 MB
  __bf16* wprojbf= (__bf16*)carve((size_t)kC * kC * 2);     //  1.2 MB
  __bf16* qkvbf  = (__bf16*)carve((size_t)kM * kC3 * 2);    // 151  MB
  __bf16* ybf    = (__bf16*)carve((size_t)kM * kC * 2);     // 50.3 MB
  float*  modln  = (float*)carve((size_t)kB * kH * kD * 4); // 24 KB

  // 1) down-convert to bf16
  {
    int n1 = kM * kC;        // 25,165,824
    int n2 = kC3 * kC;       //  1,769,472
    int n3 = kC * kC;        //    589,824
    cvt_f32_bf16<<<(n1 + 255) / 256, 256, 0, stream>>>(x, xbf, n1);
    cvt_f32_bf16<<<(n2 + 255) / 256, 256, 0, stream>>>(w_qkv, wqkvbf, n2);
    cvt_f32_bf16<<<(n3 + 255) / 256, 256, 0, stream>>>(w_proj, wprojbf, n3);
  }

  // 2) qkv = x @ w_qkv^T  -> bf16 [M, 3C]
  {
    dim3 grid(kC3 / 128, kM / 128);   // (18, 256)
    wmma_gemm_bf16<__bf16, false><<<grid, 256, 0, stream>>>(
        xbf, wqkvbf, nullptr, qkvbf, kM, kC3, kC);
  }

  // 3) modulator reduction + LayerNorm -> [B,H,D] f32
  modulator_ln_kernel<<<kB * kH, 256, 0, stream>>>(qkvbf, p2, gamma, beta, modln);

  // 4) y = q * p1 * modln -> bf16 [M, C]
  {
    size_t total = (size_t)kM * kC;
    modulate_kernel<<<(unsigned)((total + 255) / 256), 256, 0, stream>>>(
        qkvbf, p1, modln, ybf);
  }

  // 5) out = y @ w_proj^T + b_proj -> f32 [M, C]
  {
    dim3 grid(kC / 128, kM / 128);    // (6, 256)
    wmma_gemm_bf16<float, true><<<grid, 256, 0, stream>>>(
        ybf, wprojbf, b_proj, out, kM, kC, kC);
  }
}
